// SpatialBottleneck_84911503442055
// MI455X (gfx1250) — matrix operand order repaired and checked by reading, not compiled, not run on hardware
//
#include <hip/hip_runtime.h>

typedef __bf16 bf16_t;
typedef __bf16 v16bf __attribute__((ext_vector_type(16)));
typedef __bf16 v8bf  __attribute__((ext_vector_type(8)));
typedef float  v8f   __attribute__((ext_vector_type(8)));

#define HW    2500
#define WDIM  50
#define CB    256
#define CIN   1024
#define COUT  1024

#define TM    64     // macro-tile rows (M)
#define TN    128    // macro-tile cols (N = spatial)
#define TK    32     // K step (one bf16 WMMA)
#define LDSTR 40     // padded LDS pitch in bf16 (80B rows; 16B-aligned chunks, conflict-free frag reads)

#define ABUF_BYTES (TM * LDSTR * 2)   // 5120
#define BBUF_BYTES (TN * LDSTR * 2)   // 10240

// async 16B copy: global -> LDS, tracked by ASYNCcnt (no VGPR round-trip)
__device__ __forceinline__ void async_ld_b128(unsigned lds_addr, const void* gaddr) {
    asm volatile("global_load_async_to_lds_b128 %0, %1, off"
                 :: "v"(lds_addr), "v"(gaddr) : "memory");
}
__device__ __forceinline__ void wait_async0() {
    asm volatile("s_wait_asynccnt 0" ::: "memory");
}

// ---------- simple fp32 -> bf16 conversion (w1, w3) ----------
__global__ void cvt_f32_bf16(const float* __restrict__ src, bf16_t* __restrict__ dst, int n) {
    int i = blockIdx.x * blockDim.x + threadIdx.x;
    if (i < n) dst[i] = (bf16_t)src[i];
}

// ---------- w2 repack: OIHW -> [co][ (kh*3+kw)*256 + ci ] bf16 (tap-grouped K) ----------
__global__ void pack_w2(const float* __restrict__ w2, bf16_t* __restrict__ w2p) {
    int i = blockIdx.x * blockDim.x + threadIdx.x;   // over 256*2304 dst elements
    if (i < 256 * 2304) {
        int co  = i / 2304;
        int r   = i - co * 2304;
        int tap = r >> 8;          // kh*3+kw
        int ci  = r & 255;
        w2p[i] = (bf16_t)w2[(size_t)co * 2304 + ci * 9 + tap];
    }
}

// ---------- x: NCHW fp32 -> NHWC bf16, tiled transpose through LDS ----------
__global__ __launch_bounds__(256) void nchw_to_nhwc_bf16(const float* __restrict__ x,
                                                         bf16_t* __restrict__ xb) {
    __shared__ bf16_t T[64 * 80];                     // 10240 B, padded pitch
    const int n  = blockIdx.z;
    const int c0 = blockIdx.y * 64;
    const int p0 = blockIdx.x * 64;
    #pragma unroll
    for (int i = 0; i < 16; ++i) {
        int idx = threadIdx.x + 256 * i;              // 0..4095
        int c   = idx >> 6;
        int pp  = idx & 63;
        int p   = p0 + pp;
        float f = (p < HW) ? x[(size_t)(n * CIN + c0 + c) * HW + p] : 0.0f;
        T[pp * 80 + c] = (bf16_t)f;
    }
    __syncthreads();
    #pragma unroll
    for (int i = 0; i < 2; ++i) {
        int chunk = threadIdx.x + 256 * i;            // 0..511
        int pp    = chunk >> 3;
        int cc    = (chunk & 7) * 8;
        int p     = p0 + pp;
        if (p < HW) {
            v8bf v = *(const v8bf*)&T[pp * 80 + cc];
            *(v8bf*)&xb[((size_t)(n * HW) + p) * CIN + c0 + cc] = v;
        }
    }
}

// ---------- fused bottleneck GEMM stages (activations NHWC bf16) ----------
// STAGE 1: out1 = relu(s1*(w1 @ xb) + b1)                M=256,  K=1024, CA=1024
// STAGE 2: out2 = relu(s2*(w2p * out1 [3x3]) + b2)       M=256,  K=2304 (tap-grouped), CA=256
// STAGE 3: out  = relu(s3*(w3 @ out2) + b3 + x)  [NCHW]  M=1024, K=256,  CA=256
template <int STAGE>
__global__ __launch_bounds__(256)
void bottleneck_gemm(const bf16_t* __restrict__ Aw,    // bf16 weights [M][K] row-major
                     const bf16_t* __restrict__ Bact,  // NHWC bf16 activations
                     const float*  __restrict__ Xf,    // fp32 x (stage 3 residual only)
                     const float*  __restrict__ scale,
                     const float*  __restrict__ bias,
                     bf16_t*       __restrict__ Obf,   // NHWC bf16 out (stages 1,2)
                     float*        __restrict__ Of)    // NCHW fp32 out (stage 3)
{
    constexpr int K  = (STAGE == 1) ? 1024 : (STAGE == 2) ? 2304 : 256;
    constexpr int CA = (STAGE == 1) ? CIN : CB;        // NHWC channel stride of Bact
    constexpr int NK = K / TK;

    __shared__ bf16_t As[2][TM * LDSTR];               // 2 x 5120 B
    __shared__ bf16_t Bs[2][TN * LDSTR];               // 2 x 10240 B

    const int n     = blockIdx.z;
    const int m0    = blockIdx.y * TM;
    const int pbase = blockIdx.x * TN;

    const int tid  = threadIdx.x;
    const int lane = tid & 31;
    const int wid  = tid >> 5;
    const int wm   = wid & 3;            // wave's 16-row strip within 64
    const int wn   = wid >> 2;           // wave's 64-col strip within 128
    const int lr   = lane & 15;
    const bool hiH = (lane >= 16);

    // A cooperative load: one 16B chunk per thread per K-step
    const int a_row = tid >> 2;          // 0..63
    const int a_kc  = (tid & 3) * 8;     // 0,8,16,24

    // B cooperative load: 2 chunks of 8 per thread; 4 consecutive threads = 64B of one column
    int b_col[2], b_kc[2], b_pc[2], b_h[2], b_w[2];
    #pragma unroll
    for (int c = 0; c < 2; ++c) {
        int chunk = tid + c * 256;                     // 0..511
        b_col[c]  = chunk >> 2;                        // 0..127
        b_kc[c]   = (chunk & 3) * 8;                   // 0,8,16,24
        int p     = pbase + b_col[c];
        b_pc[c]   = (p < HW) ? p : (HW - 1);           // clamp: junk only lands in unstored cols
        b_h[c]    = b_pc[c] / WDIM;
        b_w[c]    = b_pc[c] - b_h[c] * WDIM;
    }

    // per-thread LDS byte offsets (flat LDS addressing uses addr[31:0])
    const unsigned aDst0 = (unsigned)(uintptr_t)&As[0][a_row * LDSTR + a_kc];
    unsigned bDst0[2];
    #pragma unroll
    for (int c = 0; c < 2; ++c)
        bDst0[c] = (unsigned)(uintptr_t)&Bs[0][b_col[c] * LDSTR + b_kc[c]];

    v8bf zerov;
    #pragma unroll
    for (int i = 0; i < 8; ++i) zerov[i] = (bf16_t)0.0f;

    // stage tile (K offset kb) into LDS buffer `buf` via async copies
    auto stage_tile = [&](int kb, int buf) {
        async_ld_b128(aDst0 + buf * ABUF_BYTES,
                      &Aw[(size_t)(m0 + a_row) * K + kb + a_kc]);
        #pragma unroll
        for (int c = 0; c < 2; ++c) {
            unsigned dst = bDst0[c] + buf * BBUF_BYTES;
            int kg0 = kb + b_kc[c];
            if (STAGE == 2) {
                int tap = kg0 >> 8;                    // kh*3+kw (K-tile never crosses a tap)
                int ci0 = kg0 & 255;
                int kh  = tap / 3;
                int kw  = tap - kh * 3;
                int hh  = b_h[c] + kh - 1;
                int ww  = b_w[c] + kw - 1;
                if (hh >= 0 && hh < WDIM && ww >= 0 && ww < WDIM) {
                    async_ld_b128(dst, &Bact[((size_t)(n * HW) + hh * WDIM + ww) * CB + ci0]);
                } else {
                    *(v8bf*)&Bs[buf][b_col[c] * LDSTR + b_kc[c]] = zerov;  // halo: zero-fill
                }
            } else {
                async_ld_b128(dst, &Bact[((size_t)(n * HW) + b_pc[c]) * CA + kg0]);
            }
        }
    };

    v8f acc[4] = {};

    // prologue: tile 0 -> buffer 0
    stage_tile(0, 0);
    wait_async0();
    __syncthreads();

    #pragma unroll 1
    for (int kt = 0; kt < NK; ++kt) {
        const int cur  = kt & 1;
        const bool more = (kt + 1 < NK);

        if (more) stage_tile((kt + 1) * TK, cur ^ 1);   // async fill of other buffer

        // ---- A fragment (ISA 16-bit A 16x32): lane<16 -> K 0..7 & 16..23; lane>=16 -> +8 ----
        const int ak0 = hiH ? 8 : 0;
        const bf16_t* ap = &As[cur][(wm * 16 + lr) * LDSTR + ak0];
        v8bf alo = *(const v8bf*)ap;
        v8bf ahi = *(const v8bf*)(ap + 16);
        v16bf afrag;
        #pragma unroll
        for (int i = 0; i < 8; ++i) { afrag[i] = alo[i]; afrag[8 + i] = ahi[i]; }

        // ---- 4 WMMAs sharing afrag; B fragment lane<16 -> K 0..7 & 16..23; lane>=16 -> +8 ----
        const int bk0 = hiH ? 8 : 0;
        #pragma unroll
        for (int j = 0; j < 4; ++j) {
            const bf16_t* bp = &Bs[cur][(wn * 64 + j * 16 + lr) * LDSTR + bk0];
            v8bf blo = *(const v8bf*)bp;
            v8bf bhi = *(const v8bf*)(bp + 16);
            v16bf bfrag;
            #pragma unroll
            for (int i = 0; i < 8; ++i) { bfrag[i] = blo[i]; bfrag[8 + i] = bhi[i]; }
            acc[j] = __builtin_amdgcn_wmma_f32_16x16x32_bf16(
                false, afrag, false, bfrag, (short)0, acc[j], false, false);
        }

        if (more) {
            wait_async0();        // my async deposits into the other buffer are done
            __syncthreads();      // publish to the workgroup (also covers the zero ds_stores)
        }
    }

    // ---- epilogue: C layout — VGPR v: M = v + (lane>=16 ? 8 : 0), N = lane%16 ----
    const int mbase = m0 + wm * 16 + (hiH ? 8 : 0);    // 8 contiguous M rows per lane
    float sc[8], bi[8];
    #pragma unroll
    for (int v = 0; v < 8; ++v) { sc[v] = scale[mbase + v]; bi[v] = bias[mbase + v]; }

    if (STAGE != 3) {
        // NHWC bf16: one 16B store per (lane, j)
        #pragma unroll
        for (int j = 0; j < 4; ++j) {
            int p = pbase + wn * 64 + j * 16 + lr;
            if (p < HW) {
                v8bf o;
                #pragma unroll
                for (int v = 0; v < 8; ++v) {
                    float r = acc[j][v] * sc[v] + bi[v];
                    o[v] = (bf16_t)(r > 0.0f ? r : 0.0f);
                }
                *(v8bf*)&Obf[((size_t)(n * HW) + p) * CB + mbase] = o;
            }
        }
    } else {
        // NCHW fp32 with residual: coalesced along p per row
        #pragma unroll
        for (int v = 0; v < 8; ++v) {
            int m = mbase + v;
            #pragma unroll
            for (int j = 0; j < 4; ++j) {
                int p = pbase + wn * 64 + j * 16 + lr;
                if (p < HW) {
                    size_t idx = (size_t)(n * COUT + m) * HW + p;
                    float r = acc[j][v] * sc[v] + bi[v] + Xf[idx];
                    Of[idx] = r > 0.0f ? r : 0.0f;
                }
            }
        }
    }
}

extern "C" void kernel_launch(void* const* d_in, const int* in_sizes, int n_in,
                              void* d_out, int out_size, void* d_ws, size_t ws_size,
                              hipStream_t stream) {
    const float* x  = (const float*)d_in[0];
    const float* w1 = (const float*)d_in[1];
    const float* w2 = (const float*)d_in[2];
    const float* w3 = (const float*)d_in[3];
    const float* s1 = (const float*)d_in[4];
    const float* b1 = (const float*)d_in[5];
    const float* s2 = (const float*)d_in[6];
    const float* b2 = (const float*)d_in[7];
    const float* s3 = (const float*)d_in[8];
    const float* b3 = (const float*)d_in[9];
    float* out = (float*)d_out;

    // workspace carve (bf16), 256B aligned
    char* ws = (char*)d_ws;
    size_t off = 0;
    auto carve = [&](size_t elems) {
        void* p = ws + off;
        off += (elems * sizeof(bf16_t) + 255) & ~(size_t)255;
        return (bf16_t*)p;
    };
    bf16_t* w1b = carve(256 * 1024);                 // [256][1024]
    bf16_t* w2b = carve(256 * 2304);                 // [256][tap*256+ci]
    bf16_t* w3b = carve(1024 * 256);                 // [1024][256]
    bf16_t* xb  = carve((size_t)32 * HW * CIN);      // x NHWC bf16
    bf16_t* o1b = carve((size_t)32 * HW * CB);       // out1 NHWC bf16
    bf16_t* o2b = carve((size_t)32 * HW * CB);       // out2 NHWC bf16

    // weight conversions / packing
    cvt_f32_bf16<<<(256 * 1024 + 255) / 256, 256, 0, stream>>>(w1, w1b, 256 * 1024);
    pack_w2<<<(256 * 2304 + 255) / 256, 256, 0, stream>>>(w2, w2b);
    cvt_f32_bf16<<<(1024 * 256 + 255) / 256, 256, 0, stream>>>(w3, w3b, 1024 * 256);

    // x -> NHWC bf16
    nchw_to_nhwc_bf16<<<dim3((HW + 63) / 64, CIN / 64, 32), 256, 0, stream>>>(x, xb);

    dim3 block(256);
    const int NT = (HW + TN - 1) / TN;  // 20

    bottleneck_gemm<1><<<dim3(NT, CB / TM, 32), block, 0, stream>>>(
        w1b, xb, nullptr, s1, b1, o1b, nullptr);
    bottleneck_gemm<2><<<dim3(NT, CB / TM, 32), block, 0, stream>>>(
        w2b, o1b, nullptr, s2, b2, o2b, nullptr);
    bottleneck_gemm<3><<<dim3(NT, COUT / TM, 32), block, 0, stream>>>(
        w3b, o2b, x, s3, b3, nullptr, out);
}
